// PhotoVoModel_47553877901560
// MI455X (gfx1250) — matrix operand order repaired and checked by reading, not compiled, not run on hardware
//
#include <hip/hip_runtime.h>
#include <hip/hip_bf16.h>

typedef __attribute__((ext_vector_type(16))) __bf16 bf16x16;
typedef __attribute__((ext_vector_type(8)))  float  v8f;
typedef __attribute__((ext_vector_type(4)))  int    v4i_t;

#define BB 32
#define MM 2048
#define NN 512
#define PP 16
#define HH 640
#define WW 640
#define DD 256
#define DIMG 384
#define NHD 8
#define PDIM 768
#define ROWS (BB * 2 * NN)   // 32768

// --------------------------------------------------------------------------
// CDNA5 async global->LDS copy (guarded; falls back to load+store)
// Probe result (round 2 stderr): builtin takes 4 args, param type is a
// generic `int ext_vector(4)*` — pass plain v4i pointers.
// --------------------------------------------------------------------------
#if defined(__gfx1250__) && \
    __has_builtin(__builtin_amdgcn_global_load_async_to_lds_b128) && \
    __has_builtin(__builtin_amdgcn_s_wait_asynccnt)
#define USE_ASYNC_LDS 1
#else
#define USE_ASYNC_LDS 0
#endif

#if USE_ASYNC_LDS
__device__ __forceinline__ void cp16(void* dstLds, const void* srcGbl) {
    __builtin_amdgcn_global_load_async_to_lds_b128(
        (v4i_t*)srcGbl, (v4i_t*)dstLds, 0, 0);
}
__device__ __forceinline__ void wait_async() {
    __builtin_amdgcn_s_wait_asynccnt(0);
}
#else
__device__ __forceinline__ void cp16(void* dstLds, const void* srcGbl) {
    *(uint4*)dstLds = *(const uint4*)srcGbl;
}
__device__ __forceinline__ void wait_async() {}
#endif

// ---------------------------------------------------------------------------
// K0: W_pe (255x768 f32) -> Wt (256x768 bf16, natural [n][k] layout, row 255 = 0)
// ---------------------------------------------------------------------------
__global__ __launch_bounds__(256) void k_conv_wpe(const float* __restrict__ W_pe,
                                                  __bf16* __restrict__ Wt) {
    int idx = blockIdx.x * 256 + threadIdx.x;     // 256*768 elements
    if (idx >= DD * PDIM) return;
    float v = (idx < (DD - 1) * PDIM) ? W_pe[idx] : 0.0f;
    Wt[idx] = (__bf16)v;
}

// ---------------------------------------------------------------------------
// K1: patch gather + clip + LayerNorm(768) -> Xn bf16 [32768,768]; score -> sc
// ---------------------------------------------------------------------------
__global__ __launch_bounds__(256) void k_patch_ln1(
    const float* __restrict__ im0, const float* __restrict__ im1,
    const float* __restrict__ kpts0, const float* __restrict__ kpts1,
    const float* __restrict__ scores0, const float* __restrict__ scores1,
    const int* __restrict__ sorted_matches,
    const float* __restrict__ ln1_g, const float* __restrict__ ln1_b,
    __bf16* __restrict__ Xn, float* __restrict__ sc) {
    int row  = blockIdx.x;            // 0..32767
    int b    = row >> 10;
    int j    = row & 1023;
    int side = j >> 9;
    int n    = j & 511;
    int tid  = threadIdx.x;

    const int* mrec = sorted_matches + ((size_t)(b * NN + n)) * 2;
    int  m     = mrec[side];
    bool valid = mrec[1] > -1;
    int  mc    = min(max(m, 0), MM - 1);

    const float* kp  = (side ? kpts1 : kpts0) + ((size_t)b * MM + mc) * 2;
    const float* scr = side ? scores1 : scores0;
    const float* img = (side ? im1 : im0) + (size_t)b * 3 * HH * WW;

    int x0 = min(max((int)rintf(kp[0]) - PP / 2, 0), WW - PP);
    int y0 = min(max((int)rintf(kp[1]) - PP / 2, 0), HH - PP);

    float v[3];
    float s1 = 0.0f, s2 = 0.0f;
#pragma unroll
    for (int t = 0; t < 3; ++t) {
        int i   = tid + t * 256;                 // i = c*256 + dy*16 + dx
        int c   = i >> 8;
        int rem = i & 255;
        int dy  = rem >> 4;
        int dx  = rem & 15;
        float val = -1.0f;
        if (valid) {
            val = img[((size_t)c * HH + (y0 + dy)) * WW + (x0 + dx)];
            val = fminf(fmaxf(val, 0.0f), 1.0f);
        }
        v[t] = val;
        s1 += val;
        s2 += val * val;
    }
    __shared__ float red0[256];
    __shared__ float red1[256];
    red0[tid] = s1; red1[tid] = s2;
    __syncthreads();
    for (int off = 128; off > 0; off >>= 1) {
        if (tid < off) { red0[tid] += red0[tid + off]; red1[tid] += red1[tid + off]; }
        __syncthreads();
    }
    float mu   = red0[0] * (1.0f / PDIM);
    float var  = red1[0] * (1.0f / PDIM) - mu * mu;
    float rstd = rsqrtf(var + 1e-5f);
#pragma unroll
    for (int t = 0; t < 3; ++t) {
        int i = tid + t * 256;
        float xn = (v[t] - mu) * rstd * ln1_g[i] + ln1_b[i];
        Xn[(size_t)row * PDIM + i] = (__bf16)xn;
    }
    if (tid == 0) sc[row] = valid ? scr[(size_t)b * MM + mc] : 0.0f;
}

// ---------------------------------------------------------------------------
// K2: WMMA GEMM  eraw[32768,256] = Xn[32768,768](bf16) @ Wt[256,768]^T + b_pe
// Block: 256 threads = 8 waves; block tile 128(M) x 128(N); wave w -> rows
// [w*16, w*16+16) x 8 N-subtiles. K loop 768/32 = 24 steps, double-buffered
// LDS with (optionally async) 16B-chunk tile copies. Both As and Bs use a
// [row][k] layout with 32-element rows, matching WMMA fragment order.
// ---------------------------------------------------------------------------
__global__ __launch_bounds__(256) void k_gemm_pe(
    const __bf16* __restrict__ Xn, const __bf16* __restrict__ Wt,
    const float* __restrict__ b_pe, float* __restrict__ eraw) {
    __shared__ __align__(16) __bf16 As[2][128 * 32];   // [m][k]
    __shared__ __align__(16) __bf16 Bs[2][128 * 32];   // [n][k]

    int tid   = threadIdx.x;
    int lane  = tid & 31;
    int w     = tid >> 5;
    int mbase = blockIdx.x * 128;
    int nbase = blockIdx.y * 128;

    v8f acc[8];
#pragma unroll
    for (int i = 0; i < 8; ++i) acc[i] = (v8f){};

    // chunk ids for this thread (512 16B chunks per tile, 2 each)
    int c0 = tid, c1 = tid + 256;
    int r0 = c0 >> 2, s0 = c0 & 3;
    int r1 = c1 >> 2, s1 = c1 & 3;

    int arow   = w * 16 + (lane & 15);
    int klane  = (lane >> 4) * 8;     // A K-group base: 0 or 8
    int bklane = (lane >> 4) * 8;     // B K-group base: 0 or 8
    int ncol   = lane & 15;

    // prologue: copy tile 0 into buffer 0
    cp16(&As[0][r0 * 32 + s0 * 8], Xn + (size_t)(mbase + r0) * PDIM + s0 * 8);
    cp16(&As[0][r1 * 32 + s1 * 8], Xn + (size_t)(mbase + r1) * PDIM + s1 * 8);
    cp16(&Bs[0][r0 * 32 + s0 * 8], Wt + (size_t)(nbase + r0) * PDIM + s0 * 8);
    cp16(&Bs[0][r1 * 32 + s1 * 8], Wt + (size_t)(nbase + r1) * PDIM + s1 * 8);

    for (int it = 0; it < PDIM / 32; ++it) {
        int buf = it & 1;
        wait_async();
        __syncthreads();
        if (it + 1 < PDIM / 32) {
            int kb = (it + 1) * 32;
            int nb = buf ^ 1;
            cp16(&As[nb][r0 * 32 + s0 * 8], Xn + (size_t)(mbase + r0) * PDIM + kb + s0 * 8);
            cp16(&As[nb][r1 * 32 + s1 * 8], Xn + (size_t)(mbase + r1) * PDIM + kb + s1 * 8);
            cp16(&Bs[nb][r0 * 32 + s0 * 8], Wt + (size_t)(nbase + r0) * PDIM + kb + s0 * 8);
            cp16(&Bs[nb][r1 * 32 + s1 * 8], Wt + (size_t)(nbase + r1) * PDIM + kb + s1 * 8);
        }

        union { unsigned int u[8]; bf16x16 v; } af;
        const unsigned int* Ap = (const unsigned int*)(&As[buf][arow * 32 + klane]);
        af.u[0] = Ap[0];  af.u[1] = Ap[1];  af.u[2] = Ap[2];  af.u[3] = Ap[3];
        af.u[4] = Ap[8];  af.u[5] = Ap[9];  af.u[6] = Ap[10]; af.u[7] = Ap[11];

#pragma unroll
        for (int ns = 0; ns < 8; ++ns) {
            union { unsigned int u[8]; bf16x16 v; } bf;
            const unsigned int* Bp =
                (const unsigned int*)(&Bs[buf][(ns * 16 + ncol) * 32 + bklane]);
#pragma unroll
            for (int i = 0; i < 8; ++i) bf.u[i] = Bp[(i & 3) + (i >> 2) * 8];
            acc[ns] = __builtin_amdgcn_wmma_f32_16x16x32_bf16(
                false, af.v, false, bf.v, (short)0, acc[ns], false, false);
        }
    }

    int moff = (lane >> 4) * 8;
#pragma unroll
    for (int ns = 0; ns < 8; ++ns) {
        int ncl = nbase + ns * 16 + ncol;
        float bias = (ncl < DD - 1) ? b_pe[ncl] : 0.0f;
#pragma unroll
        for (int r = 0; r < 8; ++r) {
            int grow = mbase + w * 16 + moff + r;
            eraw[(size_t)grow * DD + ncl] = acc[ns][r] + bias;
        }
    }
}

// ---------------------------------------------------------------------------
// K3: LayerNorm(255) in-place + append score -> pe [32768,256] f32
// ---------------------------------------------------------------------------
__global__ __launch_bounds__(256) void k_ln2_score(
    float* __restrict__ pe, const float* __restrict__ sc,
    const float* __restrict__ ln2_g, const float* __restrict__ ln2_b) {
    int row = blockIdx.x;
    int tid = threadIdx.x;
    float* prow = pe + (size_t)row * DD;
    float v = (tid < DD - 1) ? prow[tid] : 0.0f;
    __shared__ float red0[256];
    __shared__ float red1[256];
    red0[tid] = (tid < DD - 1) ? v : 0.0f;
    red1[tid] = (tid < DD - 1) ? v * v : 0.0f;
    __syncthreads();
    for (int off = 128; off > 0; off >>= 1) {
        if (tid < off) { red0[tid] += red0[tid + off]; red1[tid] += red1[tid + off]; }
        __syncthreads();
    }
    float mu   = red0[0] * (1.0f / 255.0f);
    float var  = red1[0] * (1.0f / 255.0f) - mu * mu;
    float rstd = rsqrtf(var + 1e-5f);
    if (tid < DD - 1) prow[tid] = (v - mu) * rstd * ln2_g[tid] + ln2_b[tid];
    else              prow[DD - 1] = sc[row];
}

// ---------------------------------------------------------------------------
// K4: fused per-batch attention + heads. One 256-thread block per batch.
//   att[b,h,k] = pe_row . (Wk_h^T q_h) + const_h
//   o_h        = (sum_k a_k pe_row) Wv_h^T + bv_h   (softmax weights sum to 1)
// ---------------------------------------------------------------------------
__global__ __launch_bounds__(256) void k_attn(
    const float* __restrict__ pe, const float* __restrict__ image_embs,
    const float* __restrict__ W_proj, const float* __restrict__ b_proj,
    const float* __restrict__ in_w, const float* __restrict__ in_b,
    const float* __restrict__ out_w, const float* __restrict__ out_b,
    const float* __restrict__ W_fc, const float* __restrict__ b_fc,
    float* __restrict__ out) {
    __shared__ float qv[256];
    __shared__ float qh[256];
    __shared__ float u[NHD * 256];
    __shared__ float tb[NHD];
    __shared__ float att[NHD * 1024];
    __shared__ float wsum[NHD * 256];
    __shared__ float o[256];
    __shared__ float o2[256];

    int b   = blockIdx.x;
    int tid = threadIdx.x;

    {
        const float* ie = image_embs + (size_t)b * DIMG;
        const float* wr = W_proj + (size_t)tid * DIMG;
        float acc = b_proj[tid];
        for (int c = 0; c < DIMG; ++c) acc += ie[c] * wr[c];
        qv[tid] = acc;
    }
    __syncthreads();
    {
        const float* wr = in_w + (size_t)tid * DD;
        float acc = in_b[tid];
        for (int c = 0; c < DD; ++c) acc += qv[c] * wr[c];
        qh[tid] = acc;
    }
    __syncthreads();
    {
        int c = tid;
#pragma unroll
        for (int h = 0; h < NHD; ++h) {
            float acc = 0.0f;
            for (int d = 0; d < 32; ++d)
                acc += qh[h * 32 + d] * in_w[(size_t)(DD + h * 32 + d) * DD + c];
            u[h * 256 + c] = acc;
        }
        if (tid < NHD) {
            float acc = 0.0f;
            for (int d = 0; d < 32; ++d) acc += qh[tid * 32 + d] * in_b[DD + tid * 32 + d];
            tb[tid] = acc;
        }
    }
    __syncthreads();
    const float scale = 0.17677669529663687f;  // 1/sqrt(32)
    for (int idx = tid; idx < NHD * 1024; idx += 256) {
        int h = idx >> 10;
        int k = idx & 1023;
        const float* pr = pe + ((size_t)b * 1024 + k) * DD;
        const float* ur = u + h * 256;
        float acc = tb[h];
        for (int c = 0; c < DD; ++c) acc += pr[c] * ur[c];
        att[idx] = acc * scale;
    }
    __syncthreads();
    {
        int hw   = tid >> 5;
        int lane = tid & 31;
        float m = -3.0e38f;
        for (int k = lane; k < 1024; k += 32) m = fmaxf(m, att[hw * 1024 + k]);
        for (int off = 16; off > 0; off >>= 1) m = fmaxf(m, __shfl_xor(m, off));
        float s = 0.0f;
        for (int k = lane; k < 1024; k += 32) {
            float e = expf(att[hw * 1024 + k] - m);
            att[hw * 1024 + k] = e;
            s += e;
        }
        for (int off = 16; off > 0; off >>= 1) s += __shfl_xor(s, off);
        float inv = 1.0f / s;
        for (int k = lane; k < 1024; k += 32) att[hw * 1024 + k] *= inv;
    }
    __syncthreads();
    {
        int c = tid;
        float acc[NHD] = {0, 0, 0, 0, 0, 0, 0, 0};
        for (int k = 0; k < 1024; ++k) {
            float val = pe[((size_t)b * 1024 + k) * DD + c];
#pragma unroll
            for (int h = 0; h < NHD; ++h) acc[h] += att[h * 1024 + k] * val;
        }
#pragma unroll
        for (int h = 0; h < NHD; ++h) wsum[h * 256 + c] = acc[h];
    }
    __syncthreads();
    {
        int d = tid;
        int h = d >> 5;
        const float* wv = in_w + (size_t)(2 * DD + d) * DD;
        const float* wr = wsum + h * 256;
        float acc = in_b[2 * DD + d];
        for (int c = 0; c < DD; ++c) acc += wr[c] * wv[c];
        o[d] = acc;
    }
    __syncthreads();
    {
        const float* wr = out_w + (size_t)tid * DD;
        float acc = out_b[tid];
        for (int c = 0; c < DD; ++c) acc += o[c] * wr[c];
        o2[tid] = acc;
    }
    __syncthreads();
    if (tid < 6) {
        const float* wr = W_fc + (size_t)tid * DD;
        float acc = b_fc[tid];
        for (int c = 0; c < DD; ++c) acc += o2[c] * wr[c];
        out[(size_t)b * 6 + tid] = acc;
    }
}

// ---------------------------------------------------------------------------
extern "C" void kernel_launch(void* const* d_in, const int* in_sizes, int n_in,
                              void* d_out, int out_size, void* d_ws, size_t ws_size,
                              hipStream_t stream) {
    const float* im0       = (const float*)d_in[0];
    const float* im1       = (const float*)d_in[1];
    const float* kpts0     = (const float*)d_in[2];
    const float* kpts1     = (const float*)d_in[3];
    const float* scores0   = (const float*)d_in[4];
    const float* scores1   = (const float*)d_in[5];
    const float* iembs     = (const float*)d_in[6];
    const int*   smatch    = (const int*)d_in[7];
    const float* ln1_g     = (const float*)d_in[8];
    const float* ln1_b     = (const float*)d_in[9];
    const float* W_pe      = (const float*)d_in[10];
    const float* b_pe      = (const float*)d_in[11];
    const float* ln2_g     = (const float*)d_in[12];
    const float* ln2_b     = (const float*)d_in[13];
    const float* W_proj    = (const float*)d_in[14];
    const float* b_proj    = (const float*)d_in[15];
    const float* in_w      = (const float*)d_in[16];
    const float* in_b      = (const float*)d_in[17];
    const float* out_w     = (const float*)d_in[18];
    const float* out_b     = (const float*)d_in[19];
    const float* W_fc      = (const float*)d_in[20];
    const float* b_fc      = (const float*)d_in[21];
    float* out = (float*)d_out;

    // workspace layout (all offsets 256B aligned)
    char* ws = (char*)d_ws;
    __bf16* Xn = (__bf16*)(ws + 0);                          // 32768*768*2 = 50331648
    __bf16* Wt = (__bf16*)(ws + 50331648);                   // 256*768*2   =   393216
    float*  pe = (float*)(ws + 50724864);                    // 32768*256*4 = 33554432
    float*  sc = (float*)(ws + 84279296);                    // 32768*4

    k_conv_wpe<<<(DD * PDIM) / 256, 256, 0, stream>>>(W_pe, Wt);
    k_patch_ln1<<<ROWS, 256, 0, stream>>>(im0, im1, kpts0, kpts1, scores0, scores1,
                                          smatch, ln1_g, ln1_b, Xn, sc);
    dim3 g2(ROWS / 128, DD / 128);
    k_gemm_pe<<<g2, 256, 0, stream>>>(Xn, Wt, b_pe, pe);
    k_ln2_score<<<ROWS, 256, 0, stream>>>(pe, sc, ln2_g, ln2_b);
    k_attn<<<BB, 256, 0, stream>>>(pe, iembs, W_proj, b_proj, in_w, in_b,
                                   out_w, out_b, W_fc, b_fc, out);
}
